// LocalgridAttention2d_83210696393463
// MI455X (gfx1250) — matrix operand order repaired and checked by reading, not compiled, not run on hardware
//
#include <hip/hip_runtime.h>

// ---------------------------------------------------------------------------
// Types for CDNA5 WMMA (wave32): D(f32 16x16) = A(bf16 16x32) x B(bf16 32x16) + C
// ---------------------------------------------------------------------------
typedef __attribute__((ext_vector_type(16))) __bf16 v16bf;
typedef __attribute__((ext_vector_type(8)))  __bf16 v8bf;
typedef __attribute__((ext_vector_type(8)))  float  v8f;
typedef __attribute__((ext_vector_type(8)))  unsigned short v8u;

__device__ __forceinline__ float bf2f(unsigned short u) {
    union { unsigned int i; float f; } v; v.i = ((unsigned int)u) << 16; return v.f;
}
__device__ __forceinline__ unsigned short f2bf(float x) {
    union { float f; unsigned int u; } v; v.f = x;
    unsigned int r = v.u + 0x7FFFu + ((v.u >> 16) & 1u);   // round-to-nearest-even
    return (unsigned short)(r >> 16);
}

// ---------------------------------------------------------------------------
// CDNA5 async direct-to-LDS copy (GLOBAL_LOAD_ASYNC_TO_LDS_B128, ASYNCcnt).
// Builtin signature (from toolchain diagnostics):
//   void __builtin_amdgcn_global_load_async_to_lds_b128(
//       __attribute__((address_space(1))) int __vector(4)* gsrc,   // global
//       __attribute__((address_space(3))) int __vector(4)* ldst,   // LDS
//       int imm_offset, int imm_cpol);
// ---------------------------------------------------------------------------
typedef int v4i_gcc __attribute__((__vector_size__(16)));
#define AS_GLB __attribute__((address_space(1)))
#define AS_LDS __attribute__((address_space(3)))

#if defined(__gfx1250__) && __has_builtin(__builtin_amdgcn_global_load_async_to_lds_b128)
#define HAVE_ASYNC_LDS 1
#else
#define HAVE_ASYNC_LDS 0
#endif

__device__ __forceinline__ AS_LDS v4i_gcc* to_lds(void* p) {
    // generic LDS address: low 32 bits are the LDS byte address (ISA 10.2 aperture calc)
    return (AS_LDS v4i_gcc*)(unsigned)(uintptr_t)p;
}
__device__ __forceinline__ AS_GLB v4i_gcc* to_glb(const void* p) {
    return (AS_GLB v4i_gcc*)(uintptr_t)p;   // inttoptr: sets AS(1), drops const
}

#if HAVE_ASYNC_LDS
#define ASYNC_COPY_B128(gptr, lptr) \
    __builtin_amdgcn_global_load_async_to_lds_b128(to_glb(gptr), to_lds(lptr), 0, 0)
#if __has_builtin(__builtin_amdgcn_s_wait_asynccnt)
#define WAIT_ASYNC() __builtin_amdgcn_s_wait_asynccnt(0)
#else
#define WAIT_ASYNC() asm volatile("s_wait_asynccnt 0" ::: "memory")
#endif
#endif

// ---------------------------------------------------------------------------
// fp32 -> bf16 streaming convert
// ---------------------------------------------------------------------------
__global__ void f32_to_bf16_kernel(const float* __restrict__ src,
                                   unsigned short* __restrict__ dst, long n) {
    long i = (long)blockIdx.x * blockDim.x + threadIdx.x;
    long stride = (long)gridDim.x * blockDim.x;
    for (; i < n; i += stride) dst[i] = f2bf(src[i]);
}

// src [N,K] row-major fp32 (weight, out-features x in-features)
// dst [K,N] row-major bf16 (so WMMA B-fragment rows are contiguous)
__global__ void pack_transpose_kernel(const float* __restrict__ src,
                                      unsigned short* __restrict__ dst,
                                      int N, int K) {
    long total = (long)N * K;
    long i = (long)blockIdx.x * blockDim.x + threadIdx.x;
    long stride = (long)gridDim.x * blockDim.x;
    for (; i < total; i += stride) {
        int r = (int)(i / K);     // out-feature n
        int c = (int)(i % K);     // in-feature k
        dst[(long)c * N + r] = f2bf(src[i]);
    }
}

// ---------------------------------------------------------------------------
// bf16 WMMA GEMM:  C[M,N] = A[M,K] * Bp[K,N] + bias[N]
//   Block = 256 threads = 8 waves. Block tile 256(M) x 64(N), K-step 32.
//   Wave w owns rows [w*32, w*32+32): 2 A-fragments x 4 B-fragments = 8 WMMA
//   per K-step, all fragments preloaded so one dscnt wait covers 8 WMMAs.
//   Double-buffered LDS; next tile is fetched with async direct-to-LDS DMA
//   (ASYNCcnt) while the current tile is consumed by WMMAs.
//   Requires M%256==0, N%64==0, K%32==0 (true for all calls here).
// ---------------------------------------------------------------------------
#define A_STR 40   // bf16 elems per LDS A row (80B: 16B aligned, bank-spread)
#define B_STR 72   // bf16 elems per LDS B row (144B: 16B aligned, bank-spread)

template <bool BF16_OUT>
__global__ __launch_bounds__(256)
void gemm_bf16_wmma(const __bf16* __restrict__ A, const __bf16* __restrict__ Bp,
                    const float* __restrict__ bias, void* __restrict__ Cout,
                    int M, int N, int K) {
    __shared__ __bf16 As[2][256 * A_STR];   // 2 x 20480 B
    __shared__ __bf16 Bs[2][32 * B_STR];    // 2 x  4608 B

    const int tid  = threadIdx.x;
    const int lane = tid & 31;
    const int wave = tid >> 5;
    const int half = lane >> 4;     // 0: lanes 0-15, 1: lanes 16-31
    const int l16  = lane & 15;

    const long mBase = (long)blockIdx.y * 256;
    const int  nBase = blockIdx.x * 64;

    // per-thread staging chunk coordinates
    const int brow = tid >> 3;            // B: 0..31 (K)
    const int bcol = (tid & 7) * 8;       // B: 0..56 (N)

    v8f acc[2][4] = {};                   // [m-half of wave tile][n-tile]

    // ---- prologue: stage tile k0=0 into buffer 0 ----
    {
#if HAVE_ASYNC_LDS
        #pragma unroll
        for (int it = 0; it < 4; ++it) {
            int c = tid + it * 256, row = c >> 2, col = (c & 3) * 8;
            ASYNC_COPY_B128(&A[(mBase + row) * (long)K + col], &As[0][row * A_STR + col]);
        }
        ASYNC_COPY_B128(&Bp[(long)brow * N + nBase + bcol], &Bs[0][brow * B_STR + bcol]);
        WAIT_ASYNC();
#else
        v8bf ra[4];
        #pragma unroll
        for (int it = 0; it < 4; ++it) {
            int c = tid + it * 256, row = c >> 2, col = (c & 3) * 8;
            ra[it] = *(const v8bf*)&A[(mBase + row) * (long)K + col];
        }
        v8bf rb = *(const v8bf*)&Bp[(long)brow * N + nBase + bcol];
        #pragma unroll
        for (int it = 0; it < 4; ++it) {
            int c = tid + it * 256, row = c >> 2, col = (c & 3) * 8;
            *(v8bf*)&As[0][row * A_STR + col] = ra[it];
        }
        *(v8bf*)&Bs[0][brow * B_STR + bcol] = rb;
#endif
    }
    __syncthreads();

    int cur = 0;
    for (int k0 = 0; k0 < K; k0 += 32, cur ^= 1) {
        const bool has_next = (k0 + 32) < K;
        const int  nxt = cur ^ 1;
        const int  kn  = k0 + 32;

        // ---- kick off fetch of next tile ----
#if HAVE_ASYNC_LDS
        if (has_next) {
            #pragma unroll
            for (int it = 0; it < 4; ++it) {
                int c = tid + it * 256, row = c >> 2, col = (c & 3) * 8;
                ASYNC_COPY_B128(&A[(mBase + row) * (long)K + kn + col],
                                &As[nxt][row * A_STR + col]);
            }
            ASYNC_COPY_B128(&Bp[(long)(kn + brow) * N + nBase + bcol],
                            &Bs[nxt][brow * B_STR + bcol]);
        }
#else
        v8bf ra[4]; v8bf rb;
        if (has_next) {
            #pragma unroll
            for (int it = 0; it < 4; ++it) {
                int c = tid + it * 256, row = c >> 2, col = (c & 3) * 8;
                ra[it] = *(const v8bf*)&A[(mBase + row) * (long)K + kn + col];
            }
            rb = *(const v8bf*)&Bp[(long)(kn + brow) * N + nBase + bcol];
        }
#endif

        // ---- preload all fragments of current tile, then 8 back-to-back WMMAs ----
        // A fragment (ISA 7.12.2): M = lane&15, K halves selected by lane half.
        const __bf16* ap0 = &As[cur][(wave * 32 + l16) * A_STR];
        const __bf16* ap1 = &As[cur][(wave * 32 + 16 + l16) * A_STR];
        v16bf a0 = __builtin_shufflevector(*(const v8bf*)&ap0[8 * half],
                                           *(const v8bf*)&ap0[16 + 8 * half],
            0, 1, 2, 3, 4, 5, 6, 7, 8, 9, 10, 11, 12, 13, 14, 15);
        v16bf a1 = __builtin_shufflevector(*(const v8bf*)&ap1[8 * half],
                                           *(const v8bf*)&ap1[16 + 8 * half],
            0, 1, 2, 3, 4, 5, 6, 7, 8, 9, 10, 11, 12, 13, 14, 15);

        // B fragment: column N = lane&15, K in same order as A (8*half+s, 16+8*half+s-8).
        v16bf bf[4];
        #pragma unroll
        for (int nt = 0; nt < 4; ++nt) {
            #pragma unroll
            for (int s = 0; s < 16; ++s) {
                const int kb = (s < 8) ? (8 * half + s) : (16 + 8 * half + (s - 8));
                bf[nt][s] = Bs[cur][kb * B_STR + nt * 16 + l16];
            }
        }
        #pragma unroll
        for (int nt = 0; nt < 4; ++nt) {
            acc[0][nt] = __builtin_amdgcn_wmma_f32_16x16x32_bf16(
                false, a0, false, bf[nt], (short)0, acc[0][nt], false, false);
            acc[1][nt] = __builtin_amdgcn_wmma_f32_16x16x32_bf16(
                false, a1, false, bf[nt], (short)0, acc[1][nt], false, false);
        }

        // ---- finish next-tile staging, then one barrier per K-step ----
#if HAVE_ASYNC_LDS
        if (has_next) WAIT_ASYNC();
#else
        if (has_next) {
            #pragma unroll
            for (int it = 0; it < 4; ++it) {
                int c = tid + it * 256, row = c >> 2, col = (c & 3) * 8;
                *(v8bf*)&As[nxt][row * A_STR + col] = ra[it];
            }
            *(v8bf*)&Bs[nxt][brow * B_STR + bcol] = rb;
        }
#endif
        __syncthreads();
    }

    // ---- epilogue: C/D layout N = lane&15, M = r + 8*half ----
    #pragma unroll
    for (int mh = 0; mh < 2; ++mh) {
        #pragma unroll
        for (int nt = 0; nt < 4; ++nt) {
            const int n  = nBase + nt * 16 + l16;
            const float bv = bias[n];
            #pragma unroll
            for (int r = 0; r < 8; ++r) {
                long m = mBase + wave * 32 + mh * 16 + 8 * half + r;
                float v = acc[mh][nt][r] + bv;
                if (BF16_OUT) ((unsigned short*)Cout)[m * N + n] = f2bf(v);
                else          ((float*)Cout)[m * N + n]          = v;
            }
        }
    }
}

// ---------------------------------------------------------------------------
// 7x7 local-window attention, 12 heads, hd=64.
// qkvb: bf16 [B*56*56, 2304] (q @ +0, k @ +768, v @ +1536, head*64 + d).
// One 8x8 pixel tile per (block, head, batch); 64 threads, one pixel each.
// K/V window (8+6)^2 x 64 bf16 staged in LDS (~50KB). Two-pass softmax with
// score recomputation so q[64]/out[64] stay fully in VGPRs.
// ---------------------------------------------------------------------------
__global__ __launch_bounds__(64)
void local_attn_kernel(const unsigned short* __restrict__ qkvb,
                       const float* __restrict__ rel_bias,
                       unsigned short* __restrict__ attnb) {
    const int Wt = 56, Ht = 56, HD = 64, PADW = 3;
    const int TILE = 8, LT = TILE + 2 * PADW;      // 14
    __shared__ unsigned short k_sm[LT * LT * HD];  // 14*14*64 bf16 = 25088 B
    __shared__ unsigned short v_sm[LT * LT * HD];
    __shared__ float bias_s[49];

    const int tile = blockIdx.x;       // 0..48 (7x7 tiles)
    const int head = blockIdx.y;       // 0..11
    const int b    = blockIdx.z;       // 0..15
    const int ti0  = (tile / 7) * TILE;
    const int tj0  = (tile % 7) * TILE;
    const int tid  = threadIdx.x;

    if (tid < 49) bias_s[tid] = rel_bias[head * 49 + tid];

    const long pixBase = (long)b * (Ht * Wt);

    // ---- stage K/V window (1568 x 16B chunks per matrix) ----
    for (int c = tid; c < LT * LT * 8; c += 64) {
        int p  = c >> 3;
        int d8 = (c & 7) * 8;
        int gi = ti0 - PADW + p / LT;
        int gj = tj0 - PADW + p % LT;
        v8u kk = {}, vv = {};
        if (gi >= 0 && gi < Ht && gj >= 0 && gj < Wt) {
            long off = (pixBase + (long)gi * Wt + gj) * 2304 + head * 64 + d8;
            kk = *(const v8u*)&qkvb[off + 768];
            vv = *(const v8u*)&qkvb[off + 1536];
        }
        *(v8u*)&k_sm[p * HD + d8] = kk;
        *(v8u*)&v_sm[p * HD + d8] = vv;
    }
    __syncthreads();

    const int ti = tid >> 3, tj = tid & 7;
    const int i = ti0 + ti, j = tj0 + tj;

    // ---- q into registers (fp32) ----
    float q[64];
    {
        long off = (pixBase + (long)i * Wt + j) * 2304 + head * 64;
        #pragma unroll
        for (int c = 0; c < 8; ++c) {
            v8u qq = *(const v8u*)&qkvb[off + c * 8];
            #pragma unroll
            for (int e = 0; e < 8; ++e) q[c * 8 + e] = bf2f(qq[e]);
        }
    }

    const float scale = 0.125f;        // 1/sqrt(64)
    float mx = -3.0e38f;
    for (int off = 0; off < 49; ++off) {
        int di = off / 7, dj = off % 7;
        int ni = i + di - PADW, nj = j + dj - PADW;
        if (ni < 0 || ni >= Ht || nj < 0 || nj >= Wt) continue;
        int p = (ti + di) * LT + (tj + dj);
        float s = 0.f;
        #pragma unroll
        for (int c = 0; c < 8; ++c) {
            v8u kk = *(const v8u*)&k_sm[p * HD + c * 8];
            #pragma unroll
            for (int e = 0; e < 8; ++e) s += q[c * 8 + e] * bf2f(kk[e]);
        }
        s = s * scale + bias_s[off];
        mx = fmaxf(mx, s);
    }

    float sum = 0.f;
    float out[64];
    #pragma unroll
    for (int d = 0; d < 64; ++d) out[d] = 0.f;

    for (int off = 0; off < 49; ++off) {
        int di = off / 7, dj = off % 7;
        int ni = i + di - PADW, nj = j + dj - PADW;
        if (ni < 0 || ni >= Ht || nj < 0 || nj >= Wt) continue;
        int p = (ti + di) * LT + (tj + dj);
        float s = 0.f;
        #pragma unroll
        for (int c = 0; c < 8; ++c) {
            v8u kk = *(const v8u*)&k_sm[p * HD + c * 8];
            #pragma unroll
            for (int e = 0; e < 8; ++e) s += q[c * 8 + e] * bf2f(kk[e]);
        }
        s = s * scale + bias_s[off];
        float w = __expf(s - mx);
        sum += w;
        #pragma unroll
        for (int c = 0; c < 8; ++c) {
            v8u vv = *(const v8u*)&v_sm[p * HD + c * 8];
            #pragma unroll
            for (int e = 0; e < 8; ++e) out[c * 8 + e] += w * bf2f(vv[e]);
        }
    }

    const float inv = 1.0f / sum;      // center offset always valid -> sum > 0
    long ooff = (pixBase + (long)i * Wt + j) * 768 + head * 64;
    #pragma unroll
    for (int d = 0; d < 64; ++d) attnb[ooff + d] = f2bf(out[d] * inv);
}

// ---------------------------------------------------------------------------
// Orchestration
// ---------------------------------------------------------------------------
extern "C" void kernel_launch(void* const* d_in, const int* in_sizes, int n_in,
                              void* d_out, int out_size, void* d_ws, size_t ws_size,
                              hipStream_t stream) {
    (void)in_sizes; (void)n_in; (void)out_size; (void)ws_size;
    const float* x        = (const float*)d_in[0];  // [16,56,56,768]
    const float* qkv_w    = (const float*)d_in[1];  // [2304,768]
    const float* qkv_b    = (const float*)d_in[2];  // [2304]
    const float* rel_bias = (const float*)d_in[3];  // [1,12,1,49]
    const float* proj_w   = (const float*)d_in[4];  // [768,768]
    const float* proj_b   = (const float*)d_in[5];  // [768]
    float* out = (float*)d_out;                     // [16,56,56,768] fp32

    const long Mpix = 16L * 56 * 56;                // 50176
    const int  D = 768, D3 = 2304;

    char* ws = (char*)d_ws;
    size_t off = 0;
    auto alloc = [&](size_t bytes) -> void* {
        void* p = ws + off;
        off = (off + bytes + 255) & ~(size_t)255;
        return p;
    };
    unsigned short* xb    = (unsigned short*)alloc((size_t)Mpix * D  * 2);  //  77 MB
    unsigned short* qkvb  = (unsigned short*)alloc((size_t)Mpix * D3 * 2);  // 231 MB
    unsigned short* attnb = (unsigned short*)alloc((size_t)Mpix * D  * 2);  //  77 MB
    unsigned short* wqkv  = (unsigned short*)alloc((size_t)D * D3 * 2);     // 3.5 MB
    unsigned short* wproj = (unsigned short*)alloc((size_t)D * D  * 2);     // 1.2 MB

    // 1) precision prep
    f32_to_bf16_kernel<<<2048, 256, 0, stream>>>(x, xb, Mpix * D);
    pack_transpose_kernel<<<1024, 256, 0, stream>>>(qkv_w, wqkv, D3, D);
    pack_transpose_kernel<<<512, 256, 0, stream>>>(proj_w, wproj, D, D);

    // 2) QKV GEMM: [50176,768] x [768,2304] -> bf16 qkv (+bias)
    {
        dim3 grid(D3 / 64, (unsigned)(Mpix / 256));   // 36 x 196
        gemm_bf16_wmma<true><<<grid, 256, 0, stream>>>(
            (const __bf16*)xb, (const __bf16*)wqkv, qkv_b,
            (void*)qkvb, (int)Mpix, D3, D);
    }

    // 3) local 7x7 attention
    local_attn_kernel<<<dim3(49, 12, 16), 64, 0, stream>>>(qkvb, rel_bias, attnb);

    // 4) projection GEMM: [50176,768] x [768,768] -> fp32 out (+bias)
    {
        dim3 grid(D / 64, (unsigned)(Mpix / 256));    // 12 x 196
        gemm_bf16_wmma<false><<<grid, 256, 0, stream>>>(
            (const __bf16*)attnb, (const __bf16*)wproj, proj_b,
            (void*)out, (int)Mpix, D, D);
    }
}
